// GDN_40080634806828
// MI455X (gfx1250) — matrix operand order repaired and checked by reading, not compiled, not run on hardware
//
#include <hip/hip_runtime.h>
#include <math.h>

#define NN 50000
#define EE 800000
#define INC 128
#define HID 96
#define OUTC 40
#define MTILES 3125   // 50000 / 16 exactly

typedef __attribute__((ext_vector_type(16))) __bf16 v16bf;
typedef __attribute__((ext_vector_type(8)))  float  v8f;

union BFVec { v16bf b; uint4 q[2]; };

// ---------- helpers ----------
__device__ __forceinline__ unsigned int hash2(unsigned int a, unsigned int b) {
  unsigned int h = a * 0x9E3779B9u + b * 0x85EBCA6Bu + 0x27220A95u;
  h ^= h >> 16; h *= 0x7FEB352Du;
  h ^= h >> 15; h *= 0x846CA68Bu;
  h ^= h >> 16;
  return h;
}
__device__ __forceinline__ float u01(unsigned int h) { return (float)(h >> 8) * 5.9604645e-08f; }

__device__ __forceinline__ float gauss01(unsigned int s1, unsigned int s2) {
  float u1 = fmaxf(u01(s1), 1e-7f);
  float u2 = u01(s2);
  return sqrtf(-2.0f * __logf(u1)) * __cosf(6.2831853f * u2);
}

// load 8 consecutive floats -> bf16 lane fragment half (e .. e+7)
__device__ __forceinline__ void cvt8(v16bf* dst, int ebase, const float* p) {
  float4 a0 = ((const float4*)p)[0];
  float4 a1 = ((const float4*)p)[1];
  (*dst)[ebase + 0] = (__bf16)a0.x; (*dst)[ebase + 1] = (__bf16)a0.y;
  (*dst)[ebase + 2] = (__bf16)a0.z; (*dst)[ebase + 3] = (__bf16)a0.w;
  (*dst)[ebase + 4] = (__bf16)a1.x; (*dst)[ebase + 5] = (__bf16)a1.y;
  (*dst)[ebase + 6] = (__bf16)a1.z; (*dst)[ebase + 7] = (__bf16)a1.w;
}

// mixed version: t = 0.9*a + 0.1*b
__device__ __forceinline__ void cvt8mix(v16bf* dst, int ebase, const float* pa, const float* pb) {
  float4 a0 = ((const float4*)pa)[0];
  float4 a1 = ((const float4*)pa)[1];
  float4 b0 = ((const float4*)pb)[0];
  float4 b1 = ((const float4*)pb)[1];
  (*dst)[ebase + 0] = (__bf16)(0.9f * a0.x + 0.1f * b0.x);
  (*dst)[ebase + 1] = (__bf16)(0.9f * a0.y + 0.1f * b0.y);
  (*dst)[ebase + 2] = (__bf16)(0.9f * a0.z + 0.1f * b0.z);
  (*dst)[ebase + 3] = (__bf16)(0.9f * a0.w + 0.1f * b0.w);
  (*dst)[ebase + 4] = (__bf16)(0.9f * a1.x + 0.1f * b1.x);
  (*dst)[ebase + 5] = (__bf16)(0.9f * a1.y + 0.1f * b1.y);
  (*dst)[ebase + 6] = (__bf16)(0.9f * a1.z + 0.1f * b1.z);
  (*dst)[ebase + 7] = (__bf16)(0.9f * a1.w + 0.1f * b1.w);
}

// fetch one 32x16 B chunk from transposed LDS weights: 2x ds_load_b128 per lane
__device__ __forceinline__ v16bf fetchB(const __bf16* sWt, int n, int stride, int kbase) {
  BFVec B;
  const uint4* p = (const uint4*)(sWt + n * stride + kbase);
  B.q[0] = p[0];
  B.q[1] = p[2];
  return B.b;
}

// ---------- degree / norm ----------
__global__ void deg_init(float* deg) {
  int i = blockIdx.x * blockDim.x + threadIdx.x;
  if (i < NN) deg[i] = 1.0f;                       // self loop
}
__global__ void deg_scatter(const long long* __restrict__ ei, float* deg) {
  int e = blockIdx.x * blockDim.x + threadIdx.x;
  if (e < EE) atomicAdd(&deg[(int)ei[EE + e]], 1.0f);
}
__global__ void deg_finalize(float* deg) {         // deg -> dis = rsqrt(deg)
  int i = blockIdx.x * blockDim.x + threadIdx.x;
  if (i < NN) deg[i] = rsqrtf(deg[i]);
}

// ---------- dithered quantization ----------
__global__ void quant_kernel(const float* __restrict__ hin, float* __restrict__ hq,
                             float delta, int layer) {
  int i = blockIdx.x * blockDim.x + threadIdx.x;
  if (i >= NN * HID) return;
  float u = u01(hash2(0x5EEDu + (unsigned)layer, (unsigned)i));
  float b = (u - 0.5f) * delta;
  hq[i] = floorf((hin[i] + b) / delta) * delta - b;
}

// ---------- aggregation: init with self-loop term, then edge scatter ----------
__global__ void agg_init(const float* __restrict__ hq, const float* __restrict__ dis,
                         float* __restrict__ agg) {
  int node = blockIdx.x;          // 50000 blocks of 96 threads
  int c = threadIdx.x;
  float d = dis[node];
  agg[node * HID + c] = hq[node * HID + c] * d * d;
}

__global__ __launch_bounds__(256) void edge_scatter(const long long* __restrict__ ei,
                                                    const float* __restrict__ hq,
                                                    const float* __restrict__ dis,
                                                    float* __restrict__ agg) {
  int wave = threadIdx.x >> 5;
  int lane = threadIdx.x & 31;
  int e = blockIdx.x * 8 + wave;                   // one wave per edge
  if (e >= EE) return;
  int s = (int)ei[e];
  int d = (int)ei[EE + e];
  float w = dis[s] * dis[d];
  const float* hs = hq + (long long)s * HID;
  float* ad = agg + (long long)d * HID;
#pragma unroll
  for (int j = 0; j < 3; ++j) {                    // 96 = 3 * 32, coalesced per wave
    int c = lane + j * 32;
    atomicAdd(&ad[c], hs[c] * w);
  }
}

// ---------- projection GEMM: h0 = x @ proj_w + proj_b  (M=50000,K=128,N=96) ----------
#define PROJ_KP 136   // padded column stride (u16 units), multiple of 8
__global__ __launch_bounds__(256) void proj_gemm(const float* __restrict__ x,
                                                 const float* __restrict__ w,
                                                 const float* __restrict__ bias,
                                                 float* __restrict__ h0) {
  __shared__ __bf16 sWt[HID * PROJ_KP];            // W transposed: [col][k], ~26 KB
  for (int i = threadIdx.x; i < HID * INC; i += 256) {
    int col = i >> 7, k = i & 127;
    sWt[col * PROJ_KP + k] = (__bf16)w[k * HID + col];
  }
  __syncthreads();

  int wave = threadIdx.x >> 5, lane = threadIdx.x & 31;
  int tile = blockIdx.x * 8 + wave;
  if (tile >= MTILES) return;                      // wave-uniform: EXEC stays all-1s
  int m = lane & 15, hi = lane >> 4;
  const float* xr = x + (long long)(tile * 16 + m) * INC;

  v16bf A[4];
#pragma unroll
  for (int kk = 0; kk < 4; ++kk) {                 // ISA 16-bit A 16x32 layout
    cvt8(&A[kk], 0, xr + kk * 32 + hi * 8);
    cvt8(&A[kk], 8, xr + kk * 32 + 16 + hi * 8);
  }

#pragma unroll
  for (int nt = 0; nt < 6; ++nt) {
    int n = nt * 16 + (lane & 15);
    v8f acc = {};
#pragma unroll
    for (int kk = 0; kk < 4; ++kk) {
      v16bf B = fetchB(sWt, n, PROJ_KP, kk * 32 + hi * 8);
      acc = __builtin_amdgcn_wmma_f32_16x16x32_bf16(false, A[kk], false, B,
                                                    (short)0, acc, false, false);
    }
    float bn = bias[n];
#pragma unroll
    for (int v = 0; v < 8; ++v) {
      int r = tile * 16 + v + 8 * hi;              // C/D layout: VGPR v -> row, lane -> col
      h0[(long long)r * HID + n] = acc[v] + bn;
    }
  }
}

// ---------- layer GEMM + fused epilogue ----------
// t = 0.9*agg + 0.1*h0 ; h = (1-beta)*t + beta*(t@W) ; relu(h + 0.01*N(0,1)) ; dropout
#define LAY_KP 104
__global__ __launch_bounds__(256) void layer_gemm(const float* __restrict__ agg,
                                                  const float* __restrict__ h0,
                                                  const float* __restrict__ w,
                                                  float* __restrict__ hout,
                                                  float beta, int layer, int dropout) {
  __shared__ __bf16 sWt[HID * LAY_KP];             // ~20 KB
  for (int i = threadIdx.x; i < HID * HID; i += 256) {
    int col = i / HID, k = i - col * HID;
    sWt[col * LAY_KP + k] = (__bf16)w[k * HID + col];
  }
  __syncthreads();

  int wave = threadIdx.x >> 5, lane = threadIdx.x & 31;
  int tile = blockIdx.x * 8 + wave;
  if (tile >= MTILES) return;
  int m = lane & 15, hi = lane >> 4;
  long long rowA = (long long)(tile * 16 + m) * HID;

  v16bf A[3];
#pragma unroll
  for (int kk = 0; kk < 3; ++kk) {
    cvt8mix(&A[kk], 0, agg + rowA + kk * 32 + hi * 8,      h0 + rowA + kk * 32 + hi * 8);
    cvt8mix(&A[kk], 8, agg + rowA + kk * 32 + 16 + hi * 8, h0 + rowA + kk * 32 + 16 + hi * 8);
  }

  float omb = 1.0f - beta;
#pragma unroll
  for (int nt = 0; nt < 6; ++nt) {
    int n = nt * 16 + (lane & 15);
    v8f acc = {};
#pragma unroll
    for (int kk = 0; kk < 3; ++kk) {
      v16bf B = fetchB(sWt, n, LAY_KP, kk * 32 + hi * 8);
      acc = __builtin_amdgcn_wmma_f32_16x16x32_bf16(false, A[kk], false, B,
                                                    (short)0, acc, false, false);
    }
#pragma unroll
    for (int v = 0; v < 8; ++v) {
      int r = tile * 16 + v + 8 * hi;
      unsigned int idx = (unsigned)(r * HID + n);
      float t = 0.9f * agg[idx] + 0.1f * h0[idx];  // recompute t at D layout (L2-hot)
      float val = omb * t + beta * acc[v];
      val += 0.01f * gauss01(hash2(0xA000u + (unsigned)layer, idx),
                             hash2(0xB000u + (unsigned)layer, idx));
      val = fmaxf(val, 0.0f);
      if (dropout) {
        float u = u01(hash2(0xC000u + (unsigned)layer, idx));
        val = (u < 0.5f) ? val * 2.0f : 0.0f;
      }
      hout[idx] = val;
    }
  }
}

// ---------- output GEMM: out = [h0, h] @ out_w + out_b  (K=192, N=40 padded to 48) ----------
#define OUT_KP 200
__global__ __launch_bounds__(256) void out_gemm(const float* __restrict__ h0,
                                                const float* __restrict__ h,
                                                const float* __restrict__ w,
                                                const float* __restrict__ bias,
                                                float* __restrict__ out) {
  __shared__ __bf16 sWt[48 * OUT_KP];              // zero-padded cols 40..47, ~19 KB
  for (int i = threadIdx.x; i < 48 * 192; i += 256) {
    int col = i / 192, k = i - col * 192;
    float val = (col < OUTC) ? w[k * OUTC + col] : 0.0f;
    sWt[col * OUT_KP + k] = (__bf16)val;
  }
  __syncthreads();

  int wave = threadIdx.x >> 5, lane = threadIdx.x & 31;
  int tile = blockIdx.x * 8 + wave;
  if (tile >= MTILES) return;
  int m = lane & 15, hi = lane >> 4;
  long long rowA = (long long)(tile * 16 + m) * HID;

  v16bf A[6];
#pragma unroll
  for (int kk = 0; kk < 6; ++kk) {                 // kk 0..2 from h0, 3..5 from h
    const float* src = (kk < 3) ? (h0 + rowA) : (h + rowA);
    int kb = (kk % 3) * 32;
    cvt8(&A[kk], 0, src + kb + hi * 8);
    cvt8(&A[kk], 8, src + kb + 16 + hi * 8);
  }

#pragma unroll
  for (int nt = 0; nt < 3; ++nt) {
    int n = nt * 16 + (lane & 15);
    v8f acc = {};
#pragma unroll
    for (int kk = 0; kk < 6; ++kk) {
      v16bf B = fetchB(sWt, n, OUT_KP, kk * 32 + hi * 8);
      acc = __builtin_amdgcn_wmma_f32_16x16x32_bf16(false, A[kk], false, B,
                                                    (short)0, acc, false, false);
    }
    if (n < OUTC) {
      float bn = bias[n];
#pragma unroll
      for (int v = 0; v < 8; ++v) {
        int r = tile * 16 + v + 8 * hi;
        out[(long long)r * OUTC + n] = acc[v] + bn;
      }
    }
  }
}

// ---------- host launcher ----------
extern "C" void kernel_launch(void* const* d_in, const int* in_sizes, int n_in,
                              void* d_out, int out_size, void* d_ws, size_t ws_size,
                              hipStream_t stream) {
  (void)in_sizes; (void)n_in; (void)out_size; (void)ws_size;
  const float*     x      = (const float*)d_in[0];
  const long long* eidx   = (const long long*)d_in[1];
  const float*     proj_w = (const float*)d_in[2];
  const float*     proj_b = (const float*)d_in[3];
  const float*     convws = (const float*)d_in[4];
  const float*     out_w  = (const float*)d_in[5];
  const float*     out_b  = (const float*)d_in[6];
  float*           out    = (float*)d_out;

  float* ws  = (float*)d_ws;
  float* dis = ws;                   // 50000 floats (deg, then rsqrt in-place)
  float* h0  = ws + 51200;           // 4.8M floats each, 256B aligned
  float* hcur= h0 + 4800000;
  float* hq  = hcur + 4800000;
  float* agg = hq + 4800000;

  const int NE = NN * HID;           // 4,800,000

  deg_init    <<<(NN + 255) / 256, 256, 0, stream>>>(dis);
  deg_scatter <<<(EE + 255) / 256, 256, 0, stream>>>(eidx, dis);
  deg_finalize<<<(NN + 255) / 256, 256, 0, stream>>>(dis);

  const int GEMM_GRID = (MTILES + 7) / 8;          // 8 waves/block, 1 tile/wave
  proj_gemm<<<GEMM_GRID, 256, 0, stream>>>(x, proj_w, proj_b, h0);

  const float* hin = h0;
  for (int k = 0; k < 8; ++k) {
    float delta = 1.0f / (float)(1 << k);
    quant_kernel<<<(NE + 255) / 256, 256, 0, stream>>>(hin, hq, delta, k);
    agg_init    <<<NN, HID, 0, stream>>>(hq, dis, agg);
    edge_scatter<<<(EE + 7) / 8, 256, 0, stream>>>(eidx, hq, dis, agg);
    float beta = logf(0.5f / (float)(k + 1) + 1.0f);
    layer_gemm<<<GEMM_GRID, 256, 0, stream>>>(agg, h0, convws + (size_t)k * HID * HID,
                                              hcur, beta, k, (k != 7) ? 1 : 0);
    hin = hcur;
  }

  out_gemm<<<GEMM_GRID, 256, 0, stream>>>(h0, hcur, out_w, out_b, out);
}
